// MMA_48284022342156
// MI455X (gfx1250) — matrix operand order repaired and checked by reading, not compiled, not run on hardware
//
#include <hip/hip_runtime.h>
#include <hip/hip_bf16.h>
#include <math.h>

// N=10000, DEG=32, F=128, OUT=128
#define FDIM 128
#define DEG  32

typedef __attribute__((ext_vector_type(16))) __bf16 v16bf;
typedef __attribute__((ext_vector_type(2)))  __bf16 v2bf;
typedef __attribute__((ext_vector_type(8)))  float  v8f;

// ---------------------------------------------------------------------------
// Pre-pack: fp32 -> bf16 K-pair dwords so every WMMA operand pair is one
// dword load with no per-step conversion VALU.
// ---------------------------------------------------------------------------
// Contiguous pairs (x rows: pair k,k+1 adjacent in memory).
__global__ void pack_contig_kernel(const float* __restrict__ src,
                                   v2bf* __restrict__ dst, int npairs) {
  const int i = blockIdx.x * blockDim.x + threadIdx.x;
  if (i >= npairs) return;
  const float2 v = ((const float2*)src)[i];
  v2bf p;
  p[0] = (__bf16)v.x;
  p[1] = (__bf16)v.y;
  dst[i] = p;
}

// Strided pairs (weights [K][128]: pair (k,col),(k+1,col) -> dst[k/2][col]).
__global__ void pack_strided_kernel(const float* __restrict__ src,
                                    v2bf* __restrict__ dst, int total) {
  const int i = blockIdx.x * blockDim.x + threadIdx.x;
  if (i >= total) return;
  const int kp = i >> 7, col = i & 127;
  v2bf p;
  p[0] = (__bf16)src[(2 * kp) * FDIM + col];
  p[1] = (__bf16)src[(2 * kp + 1) * FDIM + col];
  dst[i] = p;
}

// ---------------------------------------------------------------------------
// Stage 1: projections. hf=0: cen[m] = x @ Wm[0:128] (f32).
//          hf=1: y[:,m] = x @ Wm[128:256] (bf16).
// Block = 16-node tile x (m,hf); 8 waves = 8 col tiles; K=128 -> 4 WMMAs.
// ---------------------------------------------------------------------------
__global__ void __launch_bounds__(256)
proj_kernel(const v2bf* __restrict__ xpk,   // [N][64] bf16 pairs
            const v2bf* __restrict__ w0, const v2bf* __restrict__ w1,
            const v2bf* __restrict__ w2, const v2bf* __restrict__ w3,
            float* __restrict__ cen,        // [4][N][128] f32
            v2bf* __restrict__ y2,          // [N][4][64] bf16 pairs
            int N) {
  const int lane = threadIdx.x & 31;
  const int wv   = threadIdx.x >> 5;
  const int tile0 = blockIdx.x * 16;
  const int mh = blockIdx.y;
  const int m  = mh >> 1;
  const int hf = mh & 1;
  const v2bf* W = (m == 0) ? w0 : (m == 1) ? w1 : (m == 2) ? w2 : w3;

  const int half = lane >> 4;
  const int l15  = lane & 15;
  const int row  = tile0 + l15;
  const int col  = wv * 16 + l15;

  v8f acc = {};
#pragma unroll
  for (int kt = 0; kt < 4; ++kt) {
    v16bf a, b;
#pragma unroll
    for (int p = 0; p < 8; ++p) {
      const v2bf pa = xpk[(size_t)row * 64 + kt * 16 + p + ((p >= 4) ? 4 : 0) + 4 * half];
      a[2 * p] = pa[0]; a[2 * p + 1] = pa[1];
      const v2bf pb = W[(size_t)(hf * 64 + kt * 16 + p + ((p >= 4) ? 4 : 0) + 4 * half) * FDIM + col];
      b[2 * p] = pb[0]; b[2 * p + 1] = pb[1];
    }
    acc = __builtin_amdgcn_wmma_f32_16x16x32_bf16(
        false, a, false, b, (short)0, acc, false, false);
  }

  if (hf == 0) {
    float* dst = cen + (size_t)m * N * FDIM;
#pragma unroll
    for (int r = 0; r < 8; ++r) {
      const int rr = tile0 + r + 8 * half;   // C/D: M = r + 8*half
      dst[(size_t)rr * FDIM + col] = acc[r];
    }
  } else {
    __bf16* y = (__bf16*)y2;
#pragma unroll
    for (int r = 0; r < 8; ++r) {
      const int rr = tile0 + r + 8 * half;
      y[((size_t)rr * 4 + m) * FDIM + col] = (__bf16)acc[r];
    }
  }
}

// ---------------------------------------------------------------------------
// Stage 2 (bandwidth-bound bottleneck): gather + sigmoid masks + 4 aggs.
// 128 threads process 2 nodes; each thread owns a channel PAIR so the
// dominant gather stream uses b64 (x) / b32 (y) loads. Neighbor indices are
// DMA'd into LDS with global_load_async_to_lds_b32 (ASYNCcnt path).
// ---------------------------------------------------------------------------
__global__ void __launch_bounds__(128)
agg_kernel(const float* __restrict__ x,
           const int* __restrict__ nbr_idx,
           const float* __restrict__ cen,   // [4][N][128]
           const v2bf* __restrict__ y2,     // [N][4][64]
           v2bf* __restrict__ h2,           // [N][256]
           int N) {
  __shared__ int sidx[2][DEG];
  const int tid = threadIdx.x;
  const int nl  = tid >> 6;                 // node within block (0/1)
  const int pr  = tid & 63;                 // channel pair (c = 2*pr, 2*pr+1)
  const int i   = blockIdx.x * 2 + nl;

  // Async-copy 64 neighbor indices (2 nodes x 32) straight into LDS.
  if (tid < 64) {
    const int nn = tid >> 5, d = tid & 31;
    const int* gsrc = nbr_idx + (size_t)(blockIdx.x * 2 + nn) * DEG + d;
    const uint32_t lds_off = (uint32_t)(uintptr_t)&sidx[nn][d];
    asm volatile("global_load_async_to_lds_b32 %0, %1, off"
                 :: "v"(lds_off), "v"((uint64_t)(uintptr_t)gsrc)
                 : "memory");
  }
  asm volatile("s_wait_asynccnt 0" ::: "memory");
  __syncthreads();

  const float2* x2   = (const float2*)x;
  const float2* cen2 = (const float2*)cen;
  const size_t NP = (size_t)N * 64;        // pairs per cen plane

  const float2 xi = x2[(size_t)i * 64 + pr];
  const float2 c0 = cen2[0 * NP + (size_t)i * 64 + pr];
  const float2 c1 = cen2[1 * NP + (size_t)i * 64 + pr];
  const float2 c2 = cen2[2 * NP + (size_t)i * 64 + pr];
  const float2 c3 = cen2[3 * NP + (size_t)i * 64 + pr];

  float s0x = 0.f, s0y = 0.f, s1x = 0.f, s1y = 0.f;
  float mxx = -INFINITY, mxy = -INFINITY, mnx = INFINITY, mny = INFINITY;

  for (int d = 0; d < DEG; ++d) {
    const int j = sidx[nl][d];
    if (d + 1 < DEG) {                      // warm next gather rows
      const int jn = sidx[nl][d + 1];
      __builtin_prefetch(&x2[(size_t)jn * 64 + pr], 0, 1);
      __builtin_prefetch(&y2[(size_t)jn * 256 + pr], 0, 1);
    }
    const float2 xj = x2[(size_t)j * 64 + pr];
    const v2bf yv0 = y2[((size_t)j * 4 + 0) * 64 + pr];
    const v2bf yv1 = y2[((size_t)j * 4 + 1) * 64 + pr];
    const v2bf yv2 = y2[((size_t)j * 4 + 2) * 64 + pr];
    const v2bf yv3 = y2[((size_t)j * 4 + 3) * 64 + pr];

    const float g0x = 1.f / (1.f + __expf(-(c0.x + (float)yv0[0])));
    const float g0y = 1.f / (1.f + __expf(-(c0.y + (float)yv0[1])));
    const float g1x = 1.f / (1.f + __expf(-(c1.x + (float)yv1[0])));
    const float g1y = 1.f / (1.f + __expf(-(c1.y + (float)yv1[1])));
    const float g2x = 1.f / (1.f + __expf(-(c2.x + (float)yv2[0])));
    const float g2y = 1.f / (1.f + __expf(-(c2.y + (float)yv2[1])));
    const float g3x = 1.f / (1.f + __expf(-(c3.x + (float)yv3[0])));
    const float g3y = 1.f / (1.f + __expf(-(c3.y + (float)yv3[1])));

    s0x += g0x * xj.x;  s0y += g0y * xj.y;
    s1x += g1x * xj.x;  s1y += g1y * xj.y;
    mxx = fmaxf(mxx, g2x * xj.x);  mxy = fmaxf(mxy, g2y * xj.y);
    mnx = fminf(mnx, g3x * xj.x);  mny = fminf(mny, g3y * xj.y);
  }

  v2bf o;
  o[0] = (__bf16)(xi.x + s0x); o[1] = (__bf16)(xi.y + s0y);
  h2[((size_t)i * 4 + 0) * 64 + pr] = o;
  o[0] = (__bf16)(xi.x + s1x * (1.f / DEG)); o[1] = (__bf16)(xi.y + s1y * (1.f / DEG));
  h2[((size_t)i * 4 + 1) * 64 + pr] = o;
  o[0] = (__bf16)(xi.x + mxx); o[1] = (__bf16)(xi.y + mxy);
  h2[((size_t)i * 4 + 2) * 64 + pr] = o;
  o[0] = (__bf16)(xi.x + mnx); o[1] = (__bf16)(xi.y + mny);
  h2[((size_t)i * 4 + 3) * 64 + pr] = o;
}

// ---------------------------------------------------------------------------
// Stage 3: out = relu(h @ W + bias). h [N,512] bf16, Wpk pre-packed bf16.
// Block = 16-node tile; 8 waves = 8 col tiles; K=512 -> 16 WMMA steps.
// ---------------------------------------------------------------------------
__global__ void __launch_bounds__(256)
out_kernel(const v2bf* __restrict__ hpk,    // [N][256] pairs
           const v2bf* __restrict__ wfpk,   // [256][128] pairs
           const float* __restrict__ bias,  // [128]
           float* __restrict__ out,         // [N][128]
           int N) {
  const int lane = threadIdx.x & 31;
  const int wv   = threadIdx.x >> 5;
  const int tile0 = blockIdx.x * 16;
  const int half = lane >> 4;
  const int l15  = lane & 15;
  const int row  = tile0 + l15;
  const int col  = wv * 16 + l15;

  v8f acc = {};
#pragma unroll
  for (int kt = 0; kt < 16; ++kt) {
    v16bf a, b;
#pragma unroll
    for (int p = 0; p < 8; ++p) {
      const v2bf pa = hpk[(size_t)row * 256 + kt * 16 + p + ((p >= 4) ? 4 : 0) + 4 * half];
      a[2 * p] = pa[0]; a[2 * p + 1] = pa[1];
      const v2bf pb = wfpk[(size_t)(kt * 16 + p + ((p >= 4) ? 4 : 0) + 4 * half) * FDIM + col];
      b[2 * p] = pb[0]; b[2 * p + 1] = pb[1];
    }
    acc = __builtin_amdgcn_wmma_f32_16x16x32_bf16(
        false, a, false, b, (short)0, acc, false, false);
  }

  const float bc = bias[col];
#pragma unroll
  for (int r = 0; r < 8; ++r) {
    const int rr = tile0 + r + 8 * half;
    out[(size_t)rr * FDIM + col] = fmaxf(acc[r] + bc, 0.0f);
  }
}

// ---------------------------------------------------------------------------
extern "C" void kernel_launch(void* const* d_in, const int* in_sizes, int n_in,
                              void* d_out, int out_size, void* d_ws, size_t ws_size,
                              hipStream_t stream) {
  const float* x   = (const float*)d_in[0];
  const int*   nbr = (const int*)d_in[1];
  const float* m[4] = {(const float*)d_in[2], (const float*)d_in[3],
                       (const float*)d_in[4], (const float*)d_in[5]};
  const float* W   = (const float*)d_in[6];
  const float* bia = (const float*)d_in[7];
  float* out = (float*)d_out;

  const int N = in_sizes[0] / FDIM;   // 10000 (multiple of 16)
  const int ntiles = N / 16;          // 625

  // Workspace layout:
  //   cen  : 4*N*128 f32        = 20.48 MB
  //   y2   : N*4*64  v2bf       = 10.24 MB
  //   h2   : N*256   v2bf       = 10.24 MB
  //   xpk  : N*64    v2bf       =  2.56 MB
  //   wmpk : 4 * 128*128 v2bf   =  0.26 MB
  //   wfpk : 256*128 v2bf       =  0.13 MB
  char* ws = (char*)d_ws;
  float* cen = (float*)ws;                     size_t off = (size_t)4 * N * FDIM * 4;
  v2bf*  y2  = (v2bf*)(ws + off);              off += (size_t)N * 4 * 64 * 4;
  v2bf*  h2  = (v2bf*)(ws + off);              off += (size_t)N * 256 * 4;
  v2bf*  xpk = (v2bf*)(ws + off);              off += (size_t)N * 64 * 4;
  v2bf*  wmpk[4];
  for (int k = 0; k < 4; ++k) { wmpk[k] = (v2bf*)(ws + off); off += (size_t)128 * 128 * 4; }
  v2bf*  wfpk = (v2bf*)(ws + off);

  // Pre-pack operands to bf16 K-pair dwords.
  {
    const int npx = N * 64;
    pack_contig_kernel<<<(npx + 255) / 256, 256, 0, stream>>>(x, xpk, npx);
    for (int k = 0; k < 4; ++k)
      pack_strided_kernel<<<(128 * 128 + 255) / 256, 256, 0, stream>>>(m[k], wmpk[k], 128 * 128);
    pack_strided_kernel<<<(256 * 128 + 255) / 256, 256, 0, stream>>>(W, wfpk, 256 * 128);
  }

  proj_kernel<<<dim3(ntiles, 8), 256, 0, stream>>>(
      xpk, wmpk[0], wmpk[1], wmpk[2], wmpk[3], cen, y2, N);
  agg_kernel<<<(N + 1) / 2, 128, 0, stream>>>(x, nbr, cen, y2, h2, N);
  out_kernel<<<ntiles, 256, 0, stream>>>(h2, wfpk, bia, out, N);
}
